// CapacityNN_89644557402692
// MI455X (gfx1250) — matrix operand order repaired and checked by reading, not compiled, not run on hardware
//
#include <hip/hip_runtime.h>

// ---------- types ----------
typedef unsigned short u16;
typedef unsigned int   u32;
typedef __attribute__((ext_vector_type(16))) __bf16 v16bf;
typedef __attribute__((ext_vector_type(8)))  float  v8f;
typedef __attribute__((ext_vector_type(4)))  u32    u32x4;
typedef __attribute__((ext_vector_type(2)))  u32    u32x2;
typedef __attribute__((ext_vector_type(4)))  float  f32x4;
typedef int vi4 __attribute__((vector_size(16)));   // matches builtin param type

#define AS1 __attribute__((address_space(1)))
#define AS3 __attribute__((address_space(3)))

#define NPTS     64      // points per workgroup
#define HDIM     256
#define HPAD     264     // padded H row length in halves (bank-conflict free)
#define WK       64      // K-chunk for W staging
#define WKP      72      // padded W chunk row length in halves
#define NTHREADS 512     // 16 waves (wave32)

struct alignas(16) SMem {
  u16 H[2][3][NPTS][HPAD];   // ping-pong activation slabs (bf16 bits): 202,752 B
  u16 Wc[2][HDIM][WKP];      // double-buffered W K-chunks (bf16 bits):  73,728 B
  float sS[NPTS];
  float sT[NPTS];
};                            // total ~277 KB < 320 KB per WGP

static __device__ inline u16 f2bf(float f) {      // f32 -> bf16 RNE
  u32 x = __builtin_bit_cast(u32, f);
  x += 0x7FFFu + ((x >> 16) & 1u);
  return (u16)(x >> 16);
}
static __device__ inline float bf2f(u16 u) {
  u32 x = ((u32)u) << 16;
  return __builtin_bit_cast(float, x);
}

static __device__ inline v8f wmma_bf16(v16bf a, v16bf b, v8f c) {
  // D = A(16x32 bf16) * B(32x16 bf16) + C(16x16 f32)
  return __builtin_amdgcn_wmma_f32_16x16x32_bf16(
      /*neg_a=*/false, a, /*neg_b=*/false, b,
      /*c_mod=*/(short)0, c, /*reuse_a=*/false, /*reuse_b=*/false);
}

union Frag { v16bf v; u32x4 q[2]; };

// ---- 16-byte global -> LDS copy: async path if the toolchain exposes it ----
static __device__ inline void cp16_g2l(const u16* __restrict__ src, u16* dst) {
#if defined(__AMDGCN__) && __has_builtin(__builtin_amdgcn_global_load_async_to_lds_b128)
  __builtin_amdgcn_global_load_async_to_lds_b128(
      (AS1 vi4*)(const_cast<u16*>(src)), (AS3 vi4*)dst, 0, 0);
#else
  *reinterpret_cast<u32x4*>(dst) = *reinterpret_cast<const u32x4*>(src);
#endif
}
static __device__ inline void cp16_wait() {
#if defined(__AMDGCN__) && __has_builtin(__builtin_amdgcn_global_load_async_to_lds_b128)
#if __has_builtin(__builtin_amdgcn_s_wait_asynccnt)
  __builtin_amdgcn_s_wait_asynccnt(0);
#else
  asm volatile("s_wait_asynccnt 0x0" ::: "memory");
#endif
#endif
}

// ---- pre-pass: convert W1..W3 (f32, row-major [j][k]) to bf16 in d_ws ----
__global__ __launch_bounds__(256)
void convert_w_bf16(const float* __restrict__ W1, const float* __restrict__ W2,
                    const float* __restrict__ W3, u16* __restrict__ ws) {
  const int per = HDIM * HDIM;             // 65536 elements per layer
  int i = blockIdx.x * blockDim.x + threadIdx.x;   // 0..49151 (4 floats each)
  int layer = i / (per / 4);
  int off   = (i - layer * (per / 4)) * 4;
  const float* W = (layer == 0) ? W1 : (layer == 1) ? W2 : W3;
  f32x4 v = *reinterpret_cast<const f32x4*>(W + off);
  u32x2 pk;
  pk.x = ((u32)f2bf(v.y) << 16) | f2bf(v.x);
  pk.y = ((u32)f2bf(v.w) << 16) | f2bf(v.z);
  *reinterpret_cast<u32x2*>(ws + layer * per + off) = pk;
}

template <bool WSBF>
__global__ __launch_bounds__(NTHREADS)
void capnn_fused(const float* __restrict__ in,
                 const float* __restrict__ W0, const float* __restrict__ b0,
                 const float* __restrict__ W1, const float* __restrict__ b1,
                 const float* __restrict__ W2, const float* __restrict__ b2,
                 const float* __restrict__ W3, const float* __restrict__ b3,
                 const float* __restrict__ W4, const float* __restrict__ b4,
                 const float* __restrict__ lgr, const float* __restrict__ lcc,
                 const float* __restrict__ lil,
                 const float* __restrict__ imean, const float* __restrict__ istd,
                 const float* __restrict__ tmean, const float* __restrict__ tstd,
                 const u16* __restrict__ wsbf,
                 float* __restrict__ out, int N)
{
  __shared__ SMem sm;
  const int tid  = threadIdx.x;
  const int lane = tid & 31;
  const int wave = tid >> 5;
  const int hi   = lane >> 4;    // lane half: selects K sub-range of fragments
  const int ln   = lane & 15;    // M (A) / N (B,C) coordinate
  const int pt   = wave >> 2;    // point tile 0..3  -> rows pt*16..pt*16+15
  const int ch   = wave & 3;     // column group     -> tiles ch*4..ch*4+3
  const int n0   = blockIdx.x * NPTS;

  // ---- stage normalized inputs (s_norm, t_norm) ----
  if (tid < NPTS) {
    int g = n0 + tid;
    float s = 0.f, t = 0.f;
    if (g < N) { s = in[2*g]; t = in[2*g + 1]; }
    sm.sS[tid] = (s - imean[0]) / (istd[0] + 1e-8f);
    sm.sT[tid] = (t - imean[1]) / (istd[1] + 1e-8f);
  }
  __syncthreads();

  // ---- layer 0: [2] -> [256], elementwise (W0 is 256x2) ----
  // x = [s_norm, t_norm], x' = [0,1], x'' = 0  (JVP w.r.t. t_norm)
  {
    int j    = tid & 255;
    int half = tid >> 8;
    float w00 = W0[2*j], w01 = W0[2*j + 1], bj = b0[j];
    for (int n = half*32; n < half*32 + 32; ++n) {
      float zv = w00 * sm.sS[n] + w01 * sm.sT[n] + bj;
      float a  = tanhf(zv);
      float u  = 1.f - a*a;
      sm.H[0][0][n][j] = f2bf(a);                    // h
      sm.H[0][1][n][j] = f2bf(u * w01);              // h'  = u*z'
      sm.H[0][2][n][j] = f2bf(-2.f*a*u*w01*w01);     // h'' = u*z'' - 2a*u*z'^2, z''=0
    }
  }
  __syncthreads();

  // ---- 3 hidden layers: Z = H * W^T via WMMA, then tanh chain rule ----
  const v8f VZERO = {0.f,0.f,0.f,0.f,0.f,0.f,0.f,0.f};

  for (int L = 0; L < 3; ++L) {
    // ternary selects (not a pointer array!) so address-space inference keeps
    // these as global_load, not flat_load
    const float* Wf32 = (L == 0) ? W1 : (L == 1) ? W2 : W3;
    const float* bias = (L == 0) ? b1 : (L == 1) ? b2 : b3;
    const u16*   wsrc = wsbf + L * (HDIM * HDIM);   // bf16 weights (WSBF path)
    const int rb = L & 1;        // read ping-pong buffer
    const int wb = rb ^ 1;       // write buffer

    v8f acc[4][3];
    #pragma unroll
    for (int ct = 0; ct < 4; ++ct)
      #pragma unroll
      for (int st = 0; st < 3; ++st)
        acc[ct][st] = VZERO;

    for (int c = 0; c < 4; ++c) {
      // ---- stage K-chunk c of W into LDS [j][k_local] ----
      const int cb = c & 1;
      if constexpr (WSBF) {
        // pure 16B copies of pre-converted bf16 weights (async-to-LDS if available)
        #pragma unroll
        for (int it = 0; it < 4; ++it) {
          int idx = it * NTHREADS + tid;   // 2048 x 16B units per chunk
          int j   = idx >> 3;              // 8 x 16B per 64-wide row chunk
          int seg = idx & 7;
          cp16_g2l(wsrc + j*HDIM + c*WK + seg*8, &sm.Wc[cb][j][seg*8]);
        }
        if (c < 3)
          __builtin_prefetch(wsrc + (tid & 255)*HDIM + (c+1)*WK, 0, 1);
        cp16_wait();
      } else {
        // fallback: convert f32 -> bf16 while staging
        #pragma unroll
        for (int it = 0; it < 8; ++it) {
          int idx = it * NTHREADS + tid;   // float4 units: 4096 total
          int j   = idx >> 4;
          int q   = idx & 15;
          f32x4 wv = *reinterpret_cast<const f32x4*>(Wf32 + j*HDIM + c*WK + q*4);
          u32x2 pk;
          pk.x = ((u32)f2bf(wv.y) << 16) | f2bf(wv.x);
          pk.y = ((u32)f2bf(wv.w) << 16) | f2bf(wv.z);
          *reinterpret_cast<u32x2*>(&sm.Wc[cb][j][q*4]) = pk;
        }
        if (c < 3)
          __builtin_prefetch(Wf32 + (tid & 255)*HDIM + (c+1)*WK, 0, 1);
      }
      __syncthreads();

      #pragma unroll
      for (int kb = 0; kb < WK; kb += 32) {
        const int kg = c*WK + kb;
        // A fragments for the 3 Taylor streams (16x32 bf16, per-lane K pairs)
        Frag Av, Ad, Add;
        {
          int row = pt*16 + ln;
          int k0  = kg + hi*8;
          int k1  = kg + 16 + hi*8;
          Av.q[0]  = *reinterpret_cast<const u32x4*>(&sm.H[rb][0][row][k0]);
          Av.q[1]  = *reinterpret_cast<const u32x4*>(&sm.H[rb][0][row][k1]);
          Ad.q[0]  = *reinterpret_cast<const u32x4*>(&sm.H[rb][1][row][k0]);
          Ad.q[1]  = *reinterpret_cast<const u32x4*>(&sm.H[rb][1][row][k1]);
          Add.q[0] = *reinterpret_cast<const u32x4*>(&sm.H[rb][2][row][k0]);
          Add.q[1] = *reinterpret_cast<const u32x4*>(&sm.H[rb][2][row][k1]);
        }
        #pragma unroll
        for (int ct = 0; ct < 4; ++ct) {
          // B fragment (32x16): lane = N column, K split lo/hi across lane halves.
          // B[k][n] = W[n][k]  -> read from [j][k] chunk, no transpose needed.
          Frag Bf;
          int jcol = (ch*4 + ct)*16 + ln;
          int kl   = kb + hi*8;
          Bf.q[0] = *reinterpret_cast<const u32x4*>(&sm.Wc[cb][jcol][kl]);
          Bf.q[1] = *reinterpret_cast<const u32x4*>(&sm.Wc[cb][jcol][kl + 16]);
          acc[ct][0] = wmma_bf16(Av.v,  Bf.v, acc[ct][0]);  // z
          acc[ct][1] = wmma_bf16(Ad.v,  Bf.v, acc[ct][1]);  // z'
          acc[ct][2] = wmma_bf16(Add.v, Bf.v, acc[ct][2]);  // z''
        }
      }
      // no trailing barrier: next stage writes the other Wc buffer, and every
      // wave has already passed this chunk's post-stage barrier.
    }

    // tanh chain-rule epilogue directly on WMMA C layout; write other H buffer
    #pragma unroll
    for (int ct = 0; ct < 4; ++ct) {
      int jcol = (ch*4 + ct)*16 + ln;
      float bj = bias[jcol];
      #pragma unroll
      for (int i = 0; i < 8; ++i) {
        int row   = pt*16 + hi*8 + i;
        float zv  = acc[ct][0][i] + bj;
        float zd  = acc[ct][1][i];
        float zdd = acc[ct][2][i];
        float a = tanhf(zv);
        float u = 1.f - a*a;
        sm.H[wb][0][row][jcol] = f2bf(a);
        sm.H[wb][1][row][jcol] = f2bf(u * zd);
        sm.H[wb][2][row][jcol] = f2bf(u * zdd - 2.f*a*u*zd*zd);
      }
    }
    __syncthreads();
  }

  // ---- output layer: U = (h.W4 + b4)*tgt_std + tgt_mean; Verhulst residual ----
  {
    int n    = tid >> 3;       // 0..63
    int part = tid & 7;        // 8 lanes per point
    int kb   = part * 32;
    float sv = 0.f, sd = 0.f, sdd = 0.f;
    for (int k = kb; k < kb + 32; ++k) {
      float w = W4[k];
      sv  += w * bf2f(sm.H[1][0][n][k]);
      sd  += w * bf2f(sm.H[1][1][n][k]);
      sdd += w * bf2f(sm.H[1][2][n][k]);
    }
    #pragma unroll
    for (int m = 1; m < 8; m <<= 1) {
      sv  += __shfl_xor(sv,  m, 32);
      sd  += __shfl_xor(sd,  m, 32);
      sdd += __shfl_xor(sdd, m, 32);
    }
    int g = n0 + n;
    if (part == 0 && g < N) {
      float ts = tstd[0], tm = tmean[0];
      float U   = (sv + b4[0]) * ts + tm;
      float Ut  = sd  * ts;
      float Utt = sdd * ts;
      float r  = __expf(-lgr[0]);
      float Kc = 0.2f + 0.8f / (1.f + __expf(-lcc[0]));
      float Cc = 0.1f / (1.f + __expf(-lil[0]));
      float inv = 1.f / (Kc - Cc);
      float q   = (U - Cc) * inv;
      float G   = r * (U - Cc) * (1.f - q);
      float Gt  = r * Ut * (1.f - 2.f*q);
      out[g]       = U;          // U
      out[N + g]   = Ut - G;     // F
      out[2*N + g] = Utt - Gt;   // F_t
    }
  }
}

extern "C" void kernel_launch(void* const* d_in, const int* in_sizes, int n_in,
                              void* d_out, int out_size, void* d_ws, size_t ws_size,
                              hipStream_t stream) {
  const float* in   = (const float*)d_in[0];
  const float* W0   = (const float*)d_in[1];
  const float* b0   = (const float*)d_in[2];
  const float* W1   = (const float*)d_in[3];
  const float* b1   = (const float*)d_in[4];
  const float* W2   = (const float*)d_in[5];
  const float* b2   = (const float*)d_in[6];
  const float* W3   = (const float*)d_in[7];
  const float* b3   = (const float*)d_in[8];
  const float* W4   = (const float*)d_in[9];
  const float* b4   = (const float*)d_in[10];
  const float* lgr  = (const float*)d_in[11];
  const float* lcc  = (const float*)d_in[12];
  const float* lil  = (const float*)d_in[13];
  const float* imean= (const float*)d_in[14];
  const float* istd = (const float*)d_in[15];
  const float* tmean= (const float*)d_in[16];
  const float* tstd = (const float*)d_in[17];

  int N = in_sizes[0] / 2;                 // 512*128 = 65536 points
  int blocks = (N + NPTS - 1) / NPTS;      // 1024 workgroups
  const size_t ws_needed = (size_t)3 * HDIM * HDIM * sizeof(u16);  // 384 KB

  if (ws_size >= ws_needed) {
    u16* wsbf = (u16*)d_ws;
    convert_w_bf16<<<(3 * HDIM * HDIM / 4 + 255) / 256, 256, 0, stream>>>(
        W1, W2, W3, wsbf);
    capnn_fused<true><<<blocks, NTHREADS, 0, stream>>>(
        in, W0, b0, W1, b1, W2, b2, W3, b3, W4, b4,
        lgr, lcc, lil, imean, istd, tmean, tstd, wsbf,
        (float*)d_out, N);
  } else {
    capnn_fused<false><<<blocks, NTHREADS, 0, stream>>>(
        in, W0, b0, W1, b1, W2, b2, W3, b3, W4, b4,
        lgr, lcc, lil, imean, istd, tmean, tstd, (const u16*)nullptr,
        (float*)d_out, N);
  }
}
